// AS_Mamba_Block_42210938585781
// MI455X (gfx1250) — matrix operand order repaired and checked by reading, not compiled, not run on hardware
//
#include <hip/hip_runtime.h>
#include <math.h>

#define CDIV(a,b) (((a)+(b)-1)/(b))

typedef __bf16 bf16_t;
typedef bf16_t v16bf __attribute__((ext_vector_type(16)));
typedef bf16_t v8bf  __attribute__((ext_vector_type(8)));
typedef float  v8f   __attribute__((ext_vector_type(8)));

__device__ __forceinline__ float act_apply(float v, int act) {
  switch (act) {
    case 1: return v / (1.f + __expf(-v));                        // silu
    case 2: return 0.5f * v * (1.f + erff(v * 0.70710678118f));   // gelu (exact)
    case 3: return (v > 20.f) ? v : log1pf(__expf(v));            // softplus
    case 4: return 1.f / (1.f + __expf(-v));                      // sigmoid
    default: return v;
  }
}

__device__ __forceinline__ v8bf pack8(float a0, float a1, float a2, float a3,
                                      float a4, float a5, float a6, float a7) {
  v8bf o;
  o[0] = (bf16_t)a0; o[1] = (bf16_t)a1; o[2] = (bf16_t)a2; o[3] = (bf16_t)a3;
  o[4] = (bf16_t)a4; o[5] = (bf16_t)a5; o[6] = (bf16_t)a6; o[7] = (bf16_t)a7;
  return o;
}

// Branchless bounds-checked row fill: clamp indices (always-legal address),
// zero via multiplicative mask. Loads stay unconditional -> batched waits.
__device__ __forceinline__ void fill16_masked(const float* __restrict__ base,
                                              long rowBase, int stride,
                                              int k0, int fc, int K, float rowOk,
                                              float tmp[16]) {
  if (stride == 1) {
    const float* p = base + rowBase;
    #pragma unroll
    for (int j = 0; j < 16; ++j) {
      int kk = k0 + fc + j;
      int kkc = kk < K ? kk : K - 1;
      float msk = (kk < K) ? rowOk : 0.f;
      tmp[j] = p[kkc] * msk;
    }
  } else {
    #pragma unroll
    for (int j = 0; j < 16; ++j) {
      int kk = k0 + fc + j;
      int kkc = kk < K ? kk : K - 1;
      float msk = (kk < K) ? rowOk : 0.f;
      tmp[j] = base[rowBase + (long)kkc * stride] * msk;
    }
  }
}

// ---------------------------------------------------------------------------
// Generic strided GEMM: C[m,n] = act(sum_k A(m,k)*W(k,n) + bias[n]) + resid(m,n)
// A(m,k) = A[a_off + m*a_sm + k*a_sk]; W(k,n) = W[w_off + k*w_sk + n*w_sn]
// Block tile 64(M) x 64(N), K step 64. 8 waves (4x2); each wave: 16x32 slice,
// 4 x v_wmma_f32_16x16x32_bf16 per K iteration. Vectorized b128 tile staging
// for interior tiles; branchless masked fill on edges.
// ---------------------------------------------------------------------------
__global__ __launch_bounds__(256)
void k_gemm_wmma(const float* __restrict__ A, long a_off, int a_sm, int a_sk,
                 const float* __restrict__ W, long w_off, int w_sk, int w_sn,
                 const float* __restrict__ bias,
                 const float* __restrict__ resid, long r_off, int r_sm,
                 float* __restrict__ C, long c_off, int c_sm,
                 int M, int N, int K, int act)
{
  __shared__ __align__(128) bf16_t As[64][64];   // [m][k]
  __shared__ __align__(128) bf16_t Bs[64][64];   // [n][k]

  const int tid  = threadIdx.x;
  const int lane = tid & 31;
  const int wv   = tid >> 5;
  const int m0   = blockIdx.y * 64;
  const int n0   = blockIdx.x * 64;
  const int wm   = (wv >> 1) * 16;   // wave M offset (4 waves)
  const int wn   = (wv & 1) * 32;    // wave N offset (2 waves, 32 cols each)
  const int half = lane >> 4;
  const int l15  = lane & 15;

  const bool inM  = (m0 + 64 <= M);
  const bool inN  = (n0 + 64 <= N);
  const bool aFast  = inM && (a_sk == 1) && ((a_sm & 3) == 0) && ((a_off & 3) == 0);
  const bool bFastK = inN && (w_sk == 1) && ((w_sn & 3) == 0) && ((w_off & 3) == 0);
  const bool bFastN = inN && (w_sn == 1) && ((w_sk & 3) == 0) && ((w_off & 3) == 0);

  const int fr = tid >> 2;           // 0..63 (A row m / B row n)
  const int fc = (tid & 3) * 16;     // k chunk base (16 k per thread)
  const int gk = tid >> 2;           // fast-N: k index 0..63
  const int gn = (tid & 3) * 16;     // fast-N: n chunk base

  const long aRow  = a_off + (long)(m0 + fr) * a_sm;   // per-thread A row base
  const long bRowN = w_off + (long)(n0 + fr) * w_sn;   // per-thread W column base
  const long bColN = w_off + (long)(n0 + gn);          // fast-N base

  // clamped bases for masked edge fills (always-legal addresses)
  const int mmA = m0 + fr;
  const int mmAc = mmA < M ? mmA : (M - 1);
  const float aOk = (mmA < M) ? 1.f : 0.f;
  const long aRowC = a_off + (long)mmAc * a_sm;
  const int nnB = n0 + fr;
  const int nnBc = nnB < N ? nnB : (N - 1);
  const float bOk = (nnB < N) ? 1.f : 0.f;
  const long bRowC = w_off + (long)nnBc * w_sn;

  v8f acc0 = {}, acc1 = {};

  for (int k0 = 0; k0 < K; k0 += 64) {
    const bool inK = (k0 + 64 <= K);

    // ---- stage A tile ----
    if (aFast && inK) {
      const float4* p = (const float4*)(A + aRow + k0 + fc);
      #pragma unroll
      for (int q = 0; q < 2; ++q) {
        float4 u = p[2 * q], v = p[2 * q + 1];
        *(v8bf*)&As[fr][fc + 8 * q] = pack8(u.x, u.y, u.z, u.w, v.x, v.y, v.z, v.w);
      }
    } else {
      float tmp[16];
      fill16_masked(A, aRowC, a_sk, k0, fc, K, aOk, tmp);
      #pragma unroll
      for (int q = 0; q < 2; ++q)
        *(v8bf*)&As[fr][fc + 8 * q] = pack8(tmp[8*q+0], tmp[8*q+1], tmp[8*q+2], tmp[8*q+3],
                                            tmp[8*q+4], tmp[8*q+5], tmp[8*q+6], tmp[8*q+7]);
    }

    // ---- stage B tile (transposed: [n][k]) ----
    if (bFastK && inK) {
      const float4* p = (const float4*)(W + bRowN + k0 + fc);
      #pragma unroll
      for (int q = 0; q < 2; ++q) {
        float4 u = p[2 * q], v = p[2 * q + 1];
        *(v8bf*)&Bs[fr][fc + 8 * q] = pack8(u.x, u.y, u.z, u.w, v.x, v.y, v.z, v.w);
      }
    } else if (bFastN && inK) {
      const float* p = W + bColN + (long)(k0 + gk) * w_sk;
      #pragma unroll
      for (int q = 0; q < 4; ++q) {
        float4 u = *(const float4*)(p + 4 * q);
        Bs[gn + 4 * q + 0][gk] = (bf16_t)u.x;
        Bs[gn + 4 * q + 1][gk] = (bf16_t)u.y;
        Bs[gn + 4 * q + 2][gk] = (bf16_t)u.z;
        Bs[gn + 4 * q + 3][gk] = (bf16_t)u.w;
      }
    } else {
      float tmp[16];
      fill16_masked(W, bRowC, w_sk, k0, fc, K, bOk, tmp);
      #pragma unroll
      for (int q = 0; q < 2; ++q)
        *(v8bf*)&Bs[fr][fc + 8 * q] = pack8(tmp[8*q+0], tmp[8*q+1], tmp[8*q+2], tmp[8*q+3],
                                            tmp[8*q+4], tmp[8*q+5], tmp[8*q+6], tmp[8*q+7]);
    }
    __syncthreads();

    // prefetch next A tile (global_prefetch_b8)
    if (aFast && (k0 + 64 < K))
      __builtin_prefetch(A + aRow + k0 + 64 + fc, 0, 1);

    #pragma unroll
    for (int ks = 0; ks < 64; ks += 32) {
      // A fragment: lane (half,row); 16 contiguous K [ks+half*16,+16)
      union { v16bf v; v8bf h[2]; } af;
      af.h[0] = *(const v8bf*)&As[wm + l15][ks + half * 16];
      af.h[1] = *(const v8bf*)&As[wm + l15][ks + half * 16 + 8];
      // B fragments: lane (half,col); 16 contiguous K per column
      v16bf b0 = *(const v16bf*)&Bs[wn + l15][ks + half * 16];
      v16bf b1 = *(const v16bf*)&Bs[wn + 16 + l15][ks + half * 16];
      acc0 = __builtin_amdgcn_wmma_f32_16x16x32_bf16(false, af.v, false, b0,
                                                     (short)0, acc0, false, false);
      acc1 = __builtin_amdgcn_wmma_f32_16x16x32_bf16(false, af.v, false, b1,
                                                     (short)0, acc1, false, false);
    }
    __syncthreads();
  }

  // C/D layout: VGPR r, lanes 0-15 -> M=r, lanes 16-31 -> M=r+8; N = lane&15
  #pragma unroll
  for (int r = 0; r < 8; ++r) {
    int gm = m0 + wm + half * 8 + r;
    if (gm >= M) continue;
    #pragma unroll
    for (int s = 0; s < 2; ++s) {
      int gc = n0 + wn + 16 * s + l15;
      if (gc < N) {
        float v = s ? acc1[r] : acc0[r];
        if (bias) v += bias[gc];
        v = act_apply(v, act);
        if (resid) v += resid[r_off + (long)gm * r_sm + gc];
        C[c_off + (long)gm * c_sm + gc] = v;
      }
    }
  }
}

static void gemm(hipStream_t s,
                 const float* A, long aoff, int a_sm, int a_sk,
                 const float* W, long woff, int w_sk, int w_sn,
                 const float* bias, const float* resid, long roff, int rsm,
                 float* C, long coff, int csm, int M, int N, int K, int act)
{
  dim3 g(CDIV(N, 64), CDIV(M, 64));
  k_gemm_wmma<<<g, 256, 0, s>>>(A, aoff, a_sm, a_sk, W, woff, w_sk, w_sn,
                                bias, resid, roff, rsm, C, coff, csm, M, N, K, act);
}

// ---------------------------------------------------------------------------
// Auxiliary kernels
// ---------------------------------------------------------------------------
__global__ void k_pack_flow(const float* __restrict__ f0, const float* __restrict__ f1,
                            const float* __restrict__ g0, const float* __restrict__ g1,
                            float* __restrict__ dst)
{
  long idx = (long)blockIdx.x * blockDim.x + threadIdx.x;
  if (idx >= 2048L * 320) return;
  int ch = (int)(idx % 320); int tok = (int)(idx / 320);
  int img = tok >> 10, pix = tok & 1023;
  float v;
  if (ch < 256) v = (img ? f1 : f0)[ch * 1024 + pix];
  else          v = (img ? g1 : g0)[(ch - 256) * 1024 + pix];
  dst[idx] = v;
}

__global__ void k_flow_out(const float* __restrict__ F3, float* __restrict__ out)
{
  int idx = blockIdx.x * blockDim.x + threadIdx.x;
  if (idx >= 2 * 3 * 1024) return;
  int pix = idx & 1023; int q = (idx >> 10) % 3; int img = idx / (3 * 1024);
  float v = F3[((long)(img * 1024 + pix)) * 3 + q];
  if (q == 2) v = 1.f / (1.f + __expf(-v));
  out[((long)img * 323 + 320 + q) * 1024 + pix] = v;
}

__global__ void k_avgpool_pack(const float* __restrict__ f0, const float* __restrict__ f1,
                               float* __restrict__ dst)
{
  int idx = blockIdx.x * blockDim.x + threadIdx.x;
  if (idx >= 512 * 256) return;
  int c = idx & 255, m = idx >> 8;
  int s = m >> 8, l = m & 255;
  int hh = l >> 4, wh = l & 15;
  const float* f = s ? f1 : f0;
  int base = c * 1024 + (hh * 2) * 32 + wh * 2;
  dst[(long)m * 256 + c] = 0.25f * (f[base] + f[base + 1] + f[base + 32] + f[base + 33]);
}

__global__ void k_window_pack(const float* __restrict__ f0, const float* __restrict__ f1,
                              float* __restrict__ dst, int w0, int nW)
{
  long idx = (long)blockIdx.x * blockDim.x + threadIdx.x;
  if (idx >= (long)nW * 49 * 256) return;
  int c = (int)(idx & 255);
  long r = idx >> 8;
  int kk = (int)(r % 49);
  int wl = (int)(r / 49);
  int wi = w0 + wl;
  int img = wi >> 10, pix = wi & 1023;
  int py = pix >> 5, px = pix & 31;
  int yy = py + kk / 7 - 3, xx = px + kk % 7 - 3;
  float v = 0.f;
  if (yy >= 0 && yy < 32 && xx >= 0 && xx < 32)
    v = (img ? f1 : f0)[c * 1024 + yy * 32 + xx];
  dst[idx] = v;
}

__global__ __launch_bounds__(256)
void k_rmsnorm(const float* __restrict__ X, const float* __restrict__ wgt,
               float* __restrict__ Y, int M)
{
  int tok = blockIdx.x * 8 + (threadIdx.x >> 5);
  int lane = threadIdx.x & 31;
  if (tok >= M) return;
  const float* x = X + (long)tok * 256;
  float vals[8]; float ss = 0.f;
  #pragma unroll
  for (int j = 0; j < 8; ++j) { vals[j] = x[lane + 32 * j]; ss += vals[j] * vals[j]; }
  #pragma unroll
  for (int o = 16; o > 0; o >>= 1) ss += __shfl_xor(ss, o, 32);
  float r = rsqrtf(ss * (1.f / 256.f) + 1e-5f);
  float* y = Y + (long)tok * 256;
  #pragma unroll
  for (int j = 0; j < 8; ++j) y[lane + 32 * j] = vals[j] * r * wgt[lane + 32 * j];
}

__global__ void k_dwconv_silu(const float* __restrict__ XZ, const float* __restrict__ w,
                              const float* __restrict__ b, float* __restrict__ XC,
                              int M, int L)
{
  long idx = (long)blockIdx.x * blockDim.x + threadIdx.x;
  if (idx >= (long)M * 512) return;
  int m = (int)(idx >> 9), d = (int)(idx & 511);
  int t = m % L;
  float acc = b[d];
  #pragma unroll
  for (int j = 0; j < 4; ++j) {
    int tt = t - 3 + j;
    if (tt >= 0) acc += w[d * 4 + j] * XZ[(long)(m - 3 + j) * 1024 + d];
  }
  XC[(long)m * 512 + d] = acc / (1.f + __expf(-acc));
}

__global__ void k_scan(const float* __restrict__ XC, const float* __restrict__ DT,
                       const float* __restrict__ DBC, const float* __restrict__ Alog,
                       const float* __restrict__ Dp, const float* __restrict__ XZ,
                       float* __restrict__ Y, int nSeq, int L)
{
  int idx = blockIdx.x * blockDim.x + threadIdx.x;
  if (idx >= nSeq * 512) return;
  int seq = idx >> 9, d = idx & 511;
  float Av[16], h[16];
  #pragma unroll
  for (int n = 0; n < 16; ++n) { Av[n] = -__expf(Alog[d * 16 + n]); h[n] = 0.f; }
  float Dd = Dp[d];
  for (int t = 0; t < L; ++t) {
    long m = (long)seq * L + t;
    float dtv = DT[m * 512 + d];
    float xv  = XC[m * 512 + d];
    const float* bc = DBC + m * 48;
    float y = 0.f;
    #pragma unroll
    for (int n = 0; n < 16; ++n) {
      float dA = __expf(dtv * Av[n]);
      h[n] = dA * h[n] + dtv * xv * bc[16 + n];
      y += h[n] * bc[32 + n];
    }
    y += xv * Dd;
    float z = XZ[m * 1024 + 512 + d];
    Y[m * 512 + d] = y * (z / (1.f + __expf(-z)));
  }
}

__global__ void k_mean49(const float* __restrict__ src, float* __restrict__ dst,
                         int w0, int nW, int C)
{
  int idx = blockIdx.x * blockDim.x + threadIdx.x;
  if (idx >= nW * C) return;
  int wl = idx / C, c = idx % C;
  float s = 0.f;
  for (int kk = 0; kk < 49; ++kk) s += src[(long)(wl * 49 + kk) * C + c];
  dst[(long)(w0 + wl) * C + c] = s * (1.f / 49.f);
}

__global__ void k_scatter_up(const float* __restrict__ src, float* __restrict__ dst, int C)
{
  long idx = (long)blockIdx.x * blockDim.x + threadIdx.x;
  if (idx >= (long)4 * 512 * C) return;
  int o = (int)(idx % C);
  long r = idx / C;
  int m = (int)(r % 512);
  int dydx = (int)(r / 512);
  int s = m >> 8, hw = m & 255, h = hw >> 4, w = hw & 15;
  int dy = dydx >> 1, dx = dydx & 1;
  dst[((long)s * C + o) * 1024 + (2 * h + dy) * 32 + (2 * w + dx)] = src[idx];
}

__global__ void k_im2col(const float* __restrict__ src, float* __restrict__ dst,
                         int Cin, int tokmajor)
{
  long idx = (long)blockIdx.x * blockDim.x + threadIdx.x;
  if (idx >= 2048L * Cin * 9) return;
  int t9 = (int)(idx % 9);
  long r = idx / 9;
  int c = (int)(r % Cin);
  int tok = (int)(r / Cin);
  int img = tok >> 10, pix = tok & 1023;
  int y = pix >> 5, x = pix & 31;
  int yy = y + t9 / 3 - 1, xx = x + t9 % 3 - 1;
  float v = 0.f;
  if (yy >= 0 && yy < 32 && xx >= 0 && xx < 32) {
    int p2 = yy * 32 + xx;
    v = tokmajor ? src[((long)img * 1024 + p2) * Cin + c]
                 : src[((long)img * Cin + c) * 1024 + p2];
  }
  dst[idx] = v;
}

__global__ void k_bn_act(float* __restrict__ y, const float* __restrict__ g,
                         const float* __restrict__ b, int C, long total, int act)
{
  long idx = (long)blockIdx.x * blockDim.x + threadIdx.x;
  if (idx >= total) return;
  int o = (int)(idx % C);
  float s = g[o] * rsqrtf(1.f + 1e-5f);
  y[idx] = act_apply(y[idx] * s + b[o], act);
}

__global__ __launch_bounds__(256)
void k_fusion_weight(const float* __restrict__ gmu, const float* __restrict__ lm,
                     const float* __restrict__ f0, const float* __restrict__ f1,
                     const float* __restrict__ wp_w, const float* __restrict__ wp_b,
                     float* __restrict__ fused)
{
  int tok = blockIdx.x;
  int img = tok >> 10, pix = tok & 1023;
  int c = threadIdx.x;
  const float* feat = img ? f1 : f0;
  float v0 = gmu[((long)img * 256 + c) * 1024 + pix];
  float v1 = lm[(long)tok * 256 + c];
  float v2 = feat[c * 1024 + pix];
  __shared__ float red[256];
  __shared__ float logit[3];
  for (int o = 0; o < 3; ++o) {
    red[c] = v0 * wp_w[o * 768 + c] + v1 * wp_w[o * 768 + 256 + c] + v2 * wp_w[o * 768 + 512 + c];
    __syncthreads();
    for (int st = 128; st > 0; st >>= 1) { if (c < st) red[c] += red[c + st]; __syncthreads(); }
    if (c == 0) logit[o] = red[0] + wp_b[o];
    __syncthreads();
  }
  float mx = fmaxf(logit[0], fmaxf(logit[1], logit[2]));
  float e0 = __expf(logit[0] - mx), e1 = __expf(logit[1] - mx), e2 = __expf(logit[2] - mx);
  float inv = 1.f / (e0 + e1 + e2);
  fused[((long)img * 256 + c) * 1024 + pix] = (v0 * e0 + v1 * e1 + v2 * e2) * inv;
}

__global__ __launch_bounds__(256)
void k_fusion_final(const float* __restrict__ y2, const float* __restrict__ fusedBuf,
                    const float* __restrict__ g2, const float* __restrict__ b2,
                    const float* __restrict__ lng, const float* __restrict__ lnb,
                    float* __restrict__ out)
{
  int tok = blockIdx.x;
  int img = tok >> 10, pix = tok & 1023;
  int c = threadIdx.x;
  float s = g2[c] * rsqrtf(1.f + 1e-5f);
  float v = y2[(long)tok * 256 + c] * s + b2[c] + fusedBuf[((long)img * 256 + c) * 1024 + pix];
  __shared__ float red[256];
  __shared__ float mu_s, var_s;
  red[c] = v; __syncthreads();
  for (int st = 128; st > 0; st >>= 1) { if (c < st) red[c] += red[c + st]; __syncthreads(); }
  if (c == 0) mu_s = red[0] * (1.f / 256.f);
  __syncthreads();
  float d = v - mu_s;
  red[c] = d * d; __syncthreads();
  for (int st = 128; st > 0; st >>= 1) { if (c < st) red[c] += red[c + st]; __syncthreads(); }
  if (c == 0) var_s = red[0] * (1.f / 256.f);
  __syncthreads();
  out[((long)img * 323 + c) * 1024 + pix] = d * rsqrtf(var_s + 1e-5f) * lng[c] + lnb[c];
}

__global__ void k_geom_out(const float* __restrict__ ggu, const float* __restrict__ lg,
                           float* __restrict__ out)
{
  int idx = blockIdx.x * blockDim.x + threadIdx.x;
  if (idx >= 2 * 64 * 1024) return;
  int pix = idx & 1023;
  int g = (idx >> 10) & 63;
  int img = idx >> 16;
  float v = 0.5f * (ggu[((long)img * 64 + g) * 1024 + pix] +
                    lg[(long)(img * 1024 + pix) * 64 + g]);
  out[((long)img * 323 + 256 + g) * 1024 + pix] = v;
}

// ---------------------------------------------------------------------------
// Host orchestration
// ---------------------------------------------------------------------------
struct MambaP {
  const float *Alog, *D, *conv_b, *conv_w, *dt_b, *dt_w, *in_proj, *norm_w,
              *out_geom, *out_match, *x_proj;
};

static MambaP load_mp(void* const* d_in, int base) {
  MambaP p;
  p.Alog      = (const float*)d_in[base + 0];
  p.D         = (const float*)d_in[base + 1];
  p.conv_b    = (const float*)d_in[base + 2];
  p.conv_w    = (const float*)d_in[base + 3];
  p.dt_b      = (const float*)d_in[base + 4];
  p.dt_w      = (const float*)d_in[base + 5];
  p.in_proj   = (const float*)d_in[base + 6];
  p.norm_w    = (const float*)d_in[base + 7];
  p.out_geom  = (const float*)d_in[base + 8];
  p.out_match = (const float*)d_in[base + 9];
  p.x_proj    = (const float*)d_in[base + 10];
  return p;
}

static void mamba_layer_run(hipStream_t st, const MambaP& P,
                            const float* x, float* xnew, float* xn, float* xz,
                            float* xc, float* dbc, float* dt, float* ys,
                            float* geom, int M, int nSeq, int L)
{
  k_rmsnorm<<<CDIV(M, 8), 256, 0, st>>>(x, P.norm_w, xn, M);
  gemm(st, xn, 0, 256, 1, P.in_proj, 0, 1024, 1, nullptr, nullptr, 0, 0,
       xz, 0, 1024, M, 1024, 256, 0);
  {
    long tot = (long)M * 512;
    k_dwconv_silu<<<(unsigned)CDIV(tot, 256), 256, 0, st>>>(xz, P.conv_w, P.conv_b, xc, M, L);
  }
  gemm(st, xc, 0, 512, 1, P.x_proj, 0, 48, 1, nullptr, nullptr, 0, 0,
       dbc, 0, 48, M, 48, 512, 0);
  gemm(st, dbc, 0, 48, 1, P.dt_w, 0, 512, 1, P.dt_b, nullptr, 0, 0,
       dt, 0, 512, M, 512, 16, 3 /*softplus*/);
  k_scan<<<CDIV(nSeq * 512, 256), 256, 0, st>>>(xc, dt, dbc, P.Alog, P.D, xz, ys, nSeq, L);
  gemm(st, ys, 0, 512, 1, P.out_match, 0, 256, 1, nullptr, x, 0, 256,
       xnew, 0, 256, M, 256, 512, 0);
  if (geom)
    gemm(st, ys, 0, 512, 1, P.out_geom, 0, 64, 1, nullptr, nullptr, 0, 0,
         geom, 0, 64, M, 64, 512, 0);
}

extern "C" void kernel_launch(void* const* d_in, const int* in_sizes, int n_in,
                              void* d_out, int out_size, void* d_ws, size_t ws_size,
                              hipStream_t stream)
{
  (void)in_sizes; (void)n_in; (void)out_size;
  const float* feat0 = (const float*)d_in[0];
  const float* feat1 = (const float*)d_in[1];
  const float* geo0  = (const float*)d_in[2];
  const float* geo1  = (const float*)d_in[3];
  // params flattened alphabetically: flow, fusion, global[0..1], local[0..1], up
  int i = 4;
  const float* fl_b1 = (const float*)d_in[i++];
  const float* fl_b2 = (const float*)d_in[i++];
  const float* fl_b3 = (const float*)d_in[i++];
  const float* fl_w1 = (const float*)d_in[i++];   // (128,320)
  const float* fl_w2 = (const float*)d_in[i++];   // (128,128)
  const float* fl_w3 = (const float*)d_in[i++];   // (3,128)
  const float* fu_bn1_b = (const float*)d_in[i++];
  const float* fu_bn1_g = (const float*)d_in[i++];
  const float* fu_bn2_b = (const float*)d_in[i++];
  const float* fu_bn2_g = (const float*)d_in[i++];
  const float* fu_c1_b  = (const float*)d_in[i++];
  const float* fu_c1_w  = (const float*)d_in[i++]; // (512,256,3,3)
  const float* fu_c2_b  = (const float*)d_in[i++];
  const float* fu_c2_w  = (const float*)d_in[i++]; // (256,512,3,3)
  const float* fu_ln_b  = (const float*)d_in[i++];
  const float* fu_ln_g  = (const float*)d_in[i++];
  const float* fu_wp_b  = (const float*)d_in[i++];
  const float* fu_wp_w  = (const float*)d_in[i++]; // (3,768)
  int global_base = i; i += 22;
  int local_base  = i; i += 22;
  const float* up_gb = (const float*)d_in[i++];
  const float* up_gw = (const float*)d_in[i++];    // (64,64,2,2) IOHW
  const float* up_mb = (const float*)d_in[i++];
  const float* up_mw = (const float*)d_in[i++];    // (256,256,2,2) IOHW

  float* out = (float*)d_out;

  // workspace bump allocator
  size_t off = 0;
  auto alloc = [&](size_t nfl) -> float* {
    float* p = (float*)((char*)d_ws + off);
    off += ((nfl * sizeof(float) + 255) & ~(size_t)255);
    return p;
  };

  float* xf   = alloc(2048L * 320);
  float* h1   = alloc(2048L * 128);
  float* h2   = alloc(2048L * 128);
  float* f3   = alloc(2048L * 3);
  float* lm   = alloc(2048L * 256);
  float* lg   = alloc(2048L * 64);
  float* gmu  = alloc(2L * 256 * 1024);
  float* ggu  = alloc(2L * 64 * 1024);
  float* uptm = alloc(4L * 512 * 256);
  float* uptg = alloc(4L * 512 * 64);
  float* fusedBuf = alloc(2L * 256 * 1024);
  float* col1 = alloc(2048L * 2304);
  float* y1   = alloc(2048L * 512);
  float* col2 = alloc(2048L * 4608);
  float* y2   = alloc(2048L * 256);

  // size the local-window chunk from remaining workspace (deterministic)
  long availF = (ws_size > off) ? (long)((ws_size - off) / 4) : 0;
  long CW = availF / (49L * 3456);
  if (CW > 2048) CW = 2048;
  if (CW < 16) CW = 16;
  long Mm = CW * 49;

  float* bufA  = alloc(Mm * 256);
  float* bufB  = alloc(Mm * 256);
  float* xn    = alloc(Mm * 256);
  float* xz    = alloc(Mm * 1024);
  float* xc    = alloc(Mm * 512);
  float* dbc   = alloc(Mm * 64);
  float* dtb   = alloc(Mm * 512);
  float* ys    = alloc(Mm * 512);
  float* geomB = alloc(Mm * 64);

  // ---------------- Flow head ----------------
  k_pack_flow<<<(unsigned)CDIV(2048L * 320, 256), 256, 0, stream>>>(feat0, feat1, geo0, geo1, xf);
  gemm(stream, xf, 0, 320, 1, fl_w1, 0, 1, 320, fl_b1, nullptr, 0, 0,
       h1, 0, 128, 2048, 128, 320, 2 /*gelu*/);
  gemm(stream, h1, 0, 128, 1, fl_w2, 0, 1, 128, fl_b2, nullptr, 0, 0,
       h2, 0, 128, 2048, 128, 128, 2 /*gelu*/);
  gemm(stream, h2, 0, 128, 1, fl_w3, 0, 1, 128, fl_b3, nullptr, 0, 0,
       f3, 0, 3, 2048, 3, 128, 0);
  k_flow_out<<<CDIV(2 * 3 * 1024, 256), 256, 0, stream>>>(f3, out);

  // ---------------- Global Mamba (2 seqs of 256 tokens) ----------------
  k_avgpool_pack<<<CDIV(512 * 256, 256), 256, 0, stream>>>(feat0, feat1, bufA);
  {
    float* x = bufA; float* xo = bufB;
    for (int l = 0; l < 2; ++l) {
      MambaP P = load_mp(d_in, global_base + 11 * l);
      mamba_layer_run(stream, P, x, xo, xn, xz, xc, dbc, dtb, ys,
                      (l == 1) ? geomB : nullptr, 512, 2, 256);
      float* t = x; x = xo; xo = t;
    }
    // conv_transpose 2x2 stride 2 as 4 sub-pixel GEMMs; mw[i][o][dy][dx]
    for (int dydx = 0; dydx < 4; ++dydx) {
      gemm(stream, x, 0, 256, 1, up_mw, dydx, 1024, 4, up_mb, nullptr, 0, 0,
           uptm, (long)dydx * 512 * 256, 256, 512, 256, 256, 0);
      gemm(stream, geomB, 0, 64, 1, up_gw, dydx, 256, 4, up_gb, nullptr, 0, 0,
           uptg, (long)dydx * 512 * 64, 64, 512, 64, 64, 0);
    }
    k_scatter_up<<<(unsigned)CDIV(4L * 512 * 256, 256), 256, 0, stream>>>(uptm, gmu, 256);
    k_scatter_up<<<(unsigned)CDIV(4L * 512 * 64, 256), 256, 0, stream>>>(uptg, ggu, 64);
  }

  // ---------------- Local Mamba (2048 windows x 49 tokens), chunked ----------------
  for (long w0 = 0; w0 < 2048; w0 += CW) {
    int nW = (int)((2048 - w0 < CW) ? (2048 - w0) : CW);
    int M = nW * 49;
    k_window_pack<<<(unsigned)CDIV((long)M * 256, 256), 256, 0, stream>>>(feat0, feat1, bufA, (int)w0, nW);
    float* x = bufA; float* xo = bufB;
    for (int l = 0; l < 2; ++l) {
      MambaP P = load_mp(d_in, local_base + 11 * l);
      mamba_layer_run(stream, P, x, xo, xn, xz, xc, dbc, dtb, ys,
                      (l == 1) ? geomB : nullptr, M, nW, 49);
      float* t = x; x = xo; xo = t;
    }
    k_mean49<<<CDIV(nW * 256, 256), 256, 0, stream>>>(x, lm, (int)w0, nW, 256);
    k_mean49<<<CDIV(nW * 64, 256), 256, 0, stream>>>(geomB, lg, (int)w0, nW, 64);
  }

  // ---------------- Fusion ----------------
  k_fusion_weight<<<2048, 256, 0, stream>>>(gmu, lm, feat0, feat1, fu_wp_w, fu_wp_b, fusedBuf);
  k_im2col<<<(unsigned)CDIV(2048L * 256 * 9, 256), 256, 0, stream>>>(fusedBuf, col1, 256, 0);
  gemm(stream, col1, 0, 2304, 1, fu_c1_w, 0, 1, 2304, fu_c1_b, nullptr, 0, 0,
       y1, 0, 512, 2048, 512, 2304, 0);
  k_bn_act<<<(unsigned)CDIV(2048L * 512, 256), 256, 0, stream>>>(y1, fu_bn1_g, fu_bn1_b, 512, 2048L * 512, 2);
  k_im2col<<<(unsigned)CDIV(2048L * 512 * 9, 256), 256, 0, stream>>>(y1, col2, 512, 1);
  gemm(stream, col2, 0, 4608, 1, fu_c2_w, 0, 1, 4608, fu_c2_b, nullptr, 0, 0,
       y2, 0, 256, 2048, 256, 4608, 0);
  k_fusion_final<<<2048, 256, 0, stream>>>(y2, fusedBuf, fu_bn2_g, fu_bn2_b, fu_ln_g, fu_ln_b, out);

  // ---------------- Geometry output ----------------
  k_geom_out<<<CDIV(2 * 64 * 1024, 256), 256, 0, stream>>>(ggu, lg, out);
}
